// BoneMeshGAT_58944131170954
// MI455X (gfx1250) — matrix operand order repaired and checked by reading, not compiled, not run on hardware
//
#include <hip/hip_runtime.h>

// ---------------------------------------------------------------------------
// Types for CDNA5 WMMA
// ---------------------------------------------------------------------------
typedef __attribute__((ext_vector_type(16))) __bf16 v16bf;
typedef __attribute__((ext_vector_type(8)))  __bf16 v8bf;
typedef __attribute__((ext_vector_type(8)))  float  v8f;

#define NGROUP 16

// ---------------------------------------------------------------------------
// Generic helpers
// ---------------------------------------------------------------------------
__global__ void zero_f32_kernel(float* __restrict__ p, long n) {
    long i = (long)blockIdx.x * blockDim.x + threadIdx.x;
    if (i < n) p[i] = 0.0f;
}

// f32 (N x Kin) -> bf16 (N x Kp), zero padded columns Kin..Kp-1
__global__ void pad_cvt_kernel(const float* __restrict__ src,
                               __bf16* __restrict__ dst,
                               int N, int Kin, int Kp) {
    long i = (long)blockIdx.x * blockDim.x + threadIdx.x;
    if (i >= (long)N * Kp) return;
    int n = (int)(i / Kp);
    int k = (int)(i % Kp);
    dst[i] = (k < Kin) ? (__bf16)src[(long)n * Kin + k] : (__bf16)0.0f;
}

// ---------------------------------------------------------------------------
// Pack W (K x Nc, f32) into B-fragment layout (bf16), zero-padding k >= K.
// Layout: off = ((nt * kblocks + kb) * 32 + lane) * 16 + e
//   col = nt*16 + (lane&15), half = lane>>4
//   e<8  -> k = kb*32 + half*8 + e
//   e>=8 -> k = kb*32 + 16 + half*8 + (e-8)
// Each lane's 16 bf16 fragment is then one contiguous 32-byte load.
// ---------------------------------------------------------------------------
__global__ void pack_w_kernel(const float* __restrict__ W,
                              __bf16* __restrict__ Bp,
                              int K, int Nc, int kblocks) {
    long i = (long)blockIdx.x * blockDim.x + threadIdx.x;
    long total = (long)(Nc / 16) * kblocks * 32 * 16;
    if (i >= total) return;
    int e    = (int)(i & 15);
    int lane = (int)((i >> 4) & 31);
    int kb   = (int)((i >> 9) % kblocks);
    int nt   = (int)(i / ((long)kblocks << 9));
    int col  = nt * 16 + (lane & 15);
    int half = lane >> 4;
    int k    = kb * 32 + half * 8 + ((e < 8) ? e : (e + 8));
    Bp[i] = (k < K) ? (__bf16)W[(long)k * Nc + col] : (__bf16)0.0f;
}

// ---------------------------------------------------------------------------
// WMMA bf16 GEMM:  C[M x Nc] = A[M x K32] (bf16, K padded) @ Wpacked
// One wave computes a 16x64 strip: 1 A fragment reused across 4 WMMAs / k-step.
// block = (32,4)  (4 waves = 4 consecutive M tiles)
// grid  = (ceil(Mtiles/4), Nc/64)
// ---------------------------------------------------------------------------
__global__ void gemm_wmma_bf16_kernel(const __bf16* __restrict__ A,
                                      const __bf16* __restrict__ Bp,
                                      float* __restrict__ C,
                                      int Mtiles, int K32, int Nc) {
    const int lane = threadIdx.x;                            // 0..31
    const int mt   = blockIdx.x * blockDim.y + threadIdx.y;  // M tile
    const int nst  = blockIdx.y;                             // 64-col supertile
    if (mt >= Mtiles) return;                                // wave-uniform

    const int l16  = lane & 15;
    const int half = lane >> 4;
    const int kblocks = K32 >> 5;

    const __bf16* arow = A + ((long)mt * 16 + l16) * K32 + half * 8;
    const __bf16* bb   = Bp + (((long)(nst * 4) * kblocks) * 32 + lane) * 16;
    const long bstride_nt = (long)kblocks * 32 * 16;  // between n-tiles
    const long bstride_kb = 32L * 16;                 // between k-blocks

    v8f acc0 = {}, acc1 = {}, acc2 = {}, acc3 = {};

    for (int kb = 0; kb < kblocks; ++kb) {
        // A fragment: two aligned 16B loads of 8 consecutive k each
        v8bf alo = *(const v8bf*)(arow + kb * 32);
        v8bf ahi = *(const v8bf*)(arow + kb * 32 + 16);
        v16bf a;
#pragma unroll
        for (int e = 0; e < 8; ++e) { a[e] = alo[e]; a[e + 8] = ahi[e]; }

        // B fragments: one contiguous 32B load per n-tile
        const __bf16* bk = bb + kb * bstride_kb;
        v16bf b0 = *(const v16bf*)(bk + 0 * bstride_nt);
        v16bf b1 = *(const v16bf*)(bk + 1 * bstride_nt);
        v16bf b2 = *(const v16bf*)(bk + 2 * bstride_nt);
        v16bf b3 = *(const v16bf*)(bk + 3 * bstride_nt);

        acc0 = __builtin_amdgcn_wmma_f32_16x16x32_bf16(false, a, false, b0, (short)0, acc0, false, false);
        acc1 = __builtin_amdgcn_wmma_f32_16x16x32_bf16(false, a, false, b1, (short)0, acc1, false, false);
        acc2 = __builtin_amdgcn_wmma_f32_16x16x32_bf16(false, a, false, b2, (short)0, acc2, false, false);
        acc3 = __builtin_amdgcn_wmma_f32_16x16x32_bf16(false, a, false, b3, (short)0, acc3, false, false);
    }

    // Store: row = mt*16 + r + half*8, col = nst*64 + j*16 + l16
    float* crow = C + (long)(nst * 64) + l16;
#pragma unroll
    for (int r = 0; r < 8; ++r) {
        long row = (long)mt * 16 + r + half * 8;
        float* cp = crow + row * Nc;
        cp[0]  = acc0[r];
        cp[16] = acc1[r];
        cp[32] = acc2[r];
        cp[48] = acc3[r];
    }
}

// ---------------------------------------------------------------------------
// Per-node attention scalars  s_src[n,h] = h[n,h,:].a_src[h,:]  (+ zero init of
// the per-node softmax buffers)
// ---------------------------------------------------------------------------
__global__ void node_scores_kernel(const float* __restrict__ h,
                                   const float* __restrict__ a_src,
                                   const float* __restrict__ a_dst,
                                   float* __restrict__ s_src,
                                   float* __restrict__ s_dst,
                                   float* __restrict__ emax,
                                   float* __restrict__ esum,
                                   int N, int heads, int d) {
    int i = blockIdx.x * blockDim.x + threadIdx.x;
    if (i >= N * heads) return;
    int n  = i / heads;
    int hh = i - n * heads;
    const float* hp = h + (long)n * heads * d + hh * d;
    const float* as = a_src + hh * d;
    const float* ad = a_dst + hh * d;
    float ss = 0.0f, sd = 0.0f;
    for (int k = 0; k < d; ++k) {
        float v = hp[k];
        ss += v * as[k];
        sd += v * ad[k];
    }
    s_src[i] = ss;
    s_dst[i] = sd;
    emax[i]  = 0.0f;   // segment_max is clamped at 0 in the reference
    esum[i]  = 0.0f;
}

__device__ __forceinline__ float leaky02(float x) {
    return x > 0.0f ? x : 0.2f * x;
}

// Pass 1: running max of positive logits (init 0 == reference clamp)
__global__ void edge_max_kernel(const int* __restrict__ src,
                                const int* __restrict__ dst,
                                const float* __restrict__ s_src,
                                const float* __restrict__ s_dst,
                                float* __restrict__ emax,
                                int E, int heads) {
    int e = blockIdx.x * blockDim.x + threadIdx.x;
    if (e >= E) return;
    int s = src[e], t = dst[e];
    for (int h = 0; h < heads; ++h) {
        float logit = leaky02(s_src[s * heads + h] + s_dst[t * heads + h]);
        if (logit > 0.0f)
            atomicMax((unsigned int*)&emax[t * heads + h], __float_as_uint(logit));
    }
}

// Pass 2: softmax denominator
__global__ void edge_sum_kernel(const int* __restrict__ src,
                                const int* __restrict__ dst,
                                const float* __restrict__ s_src,
                                const float* __restrict__ s_dst,
                                const float* __restrict__ emax,
                                float* __restrict__ esum,
                                int E, int heads) {
    int e = blockIdx.x * blockDim.x + threadIdx.x;
    if (e >= E) return;
    int s = src[e], t = dst[e];
    for (int h = 0; h < heads; ++h) {
        float logit = leaky02(s_src[s * heads + h] + s_dst[t * heads + h]);
        float ex = __expf(logit - emax[t * heads + h]);
        atomicAdd(&esum[t * heads + h], ex);
    }
}

// Pass 3: weighted aggregation. One block per edge, one thread per feature.
__global__ void edge_aggregate_kernel(const int* __restrict__ src,
                                      const int* __restrict__ dst,
                                      const float* __restrict__ s_src,
                                      const float* __restrict__ s_dst,
                                      const float* __restrict__ emax,
                                      const float* __restrict__ esum,
                                      const float* __restrict__ h,
                                      float* __restrict__ agg,
                                      int heads, int d) {
    __shared__ float alpha[8];
    const int e = blockIdx.x;
    const int f = threadIdx.x;           // 0 .. heads*d-1
    const int s = src[e], t = dst[e];
    if (f < heads) {
        float logit = leaky02(s_src[s * heads + f] + s_dst[t * heads + f]);
        float ex    = __expf(logit - emax[t * heads + f]);
        alpha[f]    = ex / (esum[t * heads + f] + 1e-8f);
    }
    __syncthreads();
    const int hid = heads * d;
    float v = h[(long)s * hid + f] * alpha[f / d];
    atomicAdd(&agg[(long)t * hid + f], v);
}

// ---------------------------------------------------------------------------
// Column statistics (sum, sumsq) for BatchNorm.  blockDim.x == C.
// ---------------------------------------------------------------------------
__global__ void col_stats_kernel(const float* __restrict__ x,
                                 float* __restrict__ stats,
                                 int N, int C) {
    int c = threadIdx.x;
    float s = 0.0f, s2 = 0.0f;
    for (int r = blockIdx.x; r < N; r += gridDim.x) {
        float v = x[(long)r * C + c];
        s  += v;
        s2 += v * v;
    }
    atomicAdd(&stats[c], s);
    atomicAdd(&stats[C + c], s2);
}

// BN + ELU fused with the f32 -> bf16 conversion for the next WMMA GEMM.
__global__ void bn_elu_kernel(const float* __restrict__ x,
                              const float* __restrict__ stats,
                              const float* __restrict__ g,
                              const float* __restrict__ b,
                              __bf16* __restrict__ out_bf,
                              float* __restrict__ out_f32,   // may be null
                              int N, int C) {
    long i = (long)blockIdx.x * blockDim.x + threadIdx.x;
    if (i >= (long)N * C) return;
    int c = (int)(i % C);
    float mean = stats[c] / (float)N;
    float var  = stats[C + c] / (float)N - mean * mean;
    float y = (x[i] - mean) * rsqrtf(var + 1e-5f) * g[c] + b[c];
    y = (y > 0.0f) ? y : (__expf(y) - 1.0f);
    out_bf[i] = (__bf16)y;
    if (out_f32) out_f32[i] = y;
}

// ---------------------------------------------------------------------------
// Graph pooling (mean with +1 count, max clamped at 0)
// ---------------------------------------------------------------------------
__global__ void pool_kernel(const float* __restrict__ act,
                            const int* __restrict__ batch,
                            float* __restrict__ psum,
                            float* __restrict__ pmax,
                            float* __restrict__ counts,
                            int N, int C) {
    long i = (long)blockIdx.x * blockDim.x + threadIdx.x;
    if (i >= (long)N * C) return;
    int n = (int)(i / C);
    int c = (int)(i % C);
    int g = batch[n];
    float v = act[i];
    atomicAdd(&psum[g * C + c], v);
    if (v > 0.0f)
        atomicMax((unsigned int*)&pmax[g * C + c], __float_as_uint(v));
    if (c == 0) atomicAdd(&counts[g], 1.0f);
}

__global__ void gfeat_kernel(const float* __restrict__ psum,
                             const float* __restrict__ pmax,
                             const float* __restrict__ counts,
                             float* __restrict__ gfeat,
                             int G, int C) {
    int i = blockIdx.x * blockDim.x + threadIdx.x;
    if (i >= G * 2 * C) return;
    int g = i / (2 * C);
    int c = i - g * 2 * C;
    float v;
    if (c < C) v = psum[g * C + c] / (counts[g] + 1.0f);
    else       v = pmax[g * C + (c - C)];
    gfeat[i] = v;
}

// ---------------------------------------------------------------------------
// Tiny classifier head (G=16 rows) — single block, 128 threads.
// ---------------------------------------------------------------------------
__global__ void classifier_kernel(const float* __restrict__ gfeat,   // G x 128
                                  const float* __restrict__ Wc1,     // 128 x 128
                                  const float* __restrict__ bc1,
                                  const float* __restrict__ gc1,
                                  const float* __restrict__ bnc1,
                                  const float* __restrict__ Wc2,     // 128 x 64
                                  const float* __restrict__ bc2,
                                  const float* __restrict__ Wc3,     // 64 x 3
                                  const float* __restrict__ bc3,
                                  float* __restrict__ out,           // G x 3
                                  int G) {
    __shared__ float z1[NGROUP][128];
    __shared__ float z2[NGROUP][64];
    int t = threadIdx.x;  // 0..127

    // z1 = gfeat @ Wc1 + bc1  (thread t owns column t)
    for (int g = 0; g < G; ++g) {
        float acc = bc1[t];
        for (int k = 0; k < 128; ++k) acc += gfeat[g * 128 + k] * Wc1[k * 128 + t];
        z1[g][t] = acc;
    }
    // BN over G rows of own column, then ReLU
    float m = 0.0f;
    for (int g = 0; g < G; ++g) m += z1[g][t];
    m /= (float)G;
    float v = 0.0f;
    for (int g = 0; g < G; ++g) { float dd = z1[g][t] - m; v += dd * dd; }
    v /= (float)G;
    float sc = rsqrtf(v + 1e-5f) * gc1[t];
    for (int g = 0; g < G; ++g) {
        float y = (z1[g][t] - m) * sc + bnc1[t];
        z1[g][t] = y > 0.0f ? y : 0.0f;
    }
    __syncthreads();

    // z2 = relu(z1 @ Wc2 + bc2)
    if (t < 64) {
        for (int g = 0; g < G; ++g) {
            float acc = bc2[t];
            for (int k = 0; k < 128; ++k) acc += z1[g][k] * Wc2[k * 64 + t];
            z2[g][t] = acc > 0.0f ? acc : 0.0f;
        }
    }
    __syncthreads();

    // out = z2 @ Wc3 + bc3
    if (t < G * 3) {
        int g = t / 3, c = t - g * 3;
        float acc = bc3[c];
        for (int k = 0; k < 64; ++k) acc += z2[g][k] * Wc3[k * 3 + c];
        out[g * 3 + c] = acc;
    }
}

// ---------------------------------------------------------------------------
// Host driver
// ---------------------------------------------------------------------------
static inline size_t align256(size_t x) { return (x + 255) & ~(size_t)255; }

extern "C" void kernel_launch(void* const* d_in, const int* in_sizes, int n_in,
                              void* d_out, int out_size, void* d_ws, size_t ws_size,
                              hipStream_t stream) {
    (void)n_in; (void)out_size; (void)ws_size;

    const float* x     = (const float*)d_in[0];
    const int*   eidx  = (const int*)d_in[1];
    const int*   batch = (const int*)d_in[2];
    const float* W1    = (const float*)d_in[3];
    const float* as1   = (const float*)d_in[4];
    const float* ad1   = (const float*)d_in[5];
    const float* g1    = (const float*)d_in[6];
    const float* b1    = (const float*)d_in[7];
    const float* W2    = (const float*)d_in[8];
    const float* as2   = (const float*)d_in[9];
    const float* ad2   = (const float*)d_in[10];
    const float* g2    = (const float*)d_in[11];
    const float* b2    = (const float*)d_in[12];
    const float* W3    = (const float*)d_in[13];
    const float* as3   = (const float*)d_in[14];
    const float* ad3   = (const float*)d_in[15];
    const float* g3    = (const float*)d_in[16];
    const float* b3    = (const float*)d_in[17];
    const float* Wc1   = (const float*)d_in[18];
    const float* bc1   = (const float*)d_in[19];
    const float* gc1   = (const float*)d_in[20];
    const float* bnc1  = (const float*)d_in[21];
    const float* Wc2   = (const float*)d_in[22];
    const float* bc2   = (const float*)d_in[23];
    const float* Wc3   = (const float*)d_in[24];
    const float* bc3   = (const float*)d_in[25];

    const int N = in_sizes[0] / 4;       // 100000
    const int E = in_sizes[1] / 2;       // 1000000
    const int G = NGROUP;
    const int HID = 256, D = 64;

    const int* src = eidx;
    const int* dst = eidx + E;

    // ---- workspace layout ----
    char* w = (char*)d_ws;
    size_t off = 0;
    auto walloc = [&](size_t bytes) -> void* {
        void* p = w + off;
        off = align256(off + bytes);
        return p;
    };
    float*  h_buf   = (float*)walloc((size_t)N * HID * 4);   // GEMM output (per layer)
    float*  agg_buf = (float*)walloc((size_t)N * HID * 4);   // aggregation output
    __bf16* actbf   = (__bf16*)walloc((size_t)N * HID * 2);  // bf16 activations (GEMM A)
    __bf16* wpack   = (__bf16*)walloc((size_t)HID * HID * 2);// packed bf16 weights (GEMM B)
    float*  ssrc    = (float*)walloc((size_t)N * 4 * 4);
    float*  sdst    = (float*)walloc((size_t)N * 4 * 4);
    float*  emax    = (float*)walloc((size_t)N * 4 * 4);
    float*  esum    = (float*)walloc((size_t)N * 4 * 4);
    float*  stats   = (float*)walloc(2 * HID * 4);
    float*  counts  = (float*)walloc(G * 4);
    float*  psum    = (float*)walloc((size_t)G * D * 4);
    float*  pmax    = (float*)walloc((size_t)G * D * 4);
    float*  gfeat   = (float*)walloc((size_t)G * 2 * D * 4);
    float*  act3    = h_buf;   // reuse: h_buf is free by the time BN of layer 3 runs

    float* out = (float*)d_out;

    auto launch_zero = [&](float* p, long n) {
        int blocks = (int)((n + 255) / 256);
        zero_f32_kernel<<<blocks, 256, 0, stream>>>(p, n);
    };

    // Runs one GAT layer. bf16 input already in `actbf` (N x K32, K zero-padded).
    auto run_layer = [&](const float* Wraw, int K, int K32, int Cout,
                         int heads, int d,
                         const float* a_s, const float* a_d,
                         const float* bn_g, const float* bn_b,
                         float* bn_f32_out) {
        const int kblocks = K32 / 32;
        // Pack weights into B-fragment layout (handles k >= K zero pad).
        long ptot = (long)(Cout / 16) * kblocks * 512;
        pack_w_kernel<<<(int)((ptot + 255) / 256), 256, 0, stream>>>(
            Wraw, wpack, K, Cout, kblocks);

        // GEMM (WMMA bf16): h = actbf @ W
        const int Mtiles = N / 16;
        dim3 gblk(32, 4);
        dim3 ggrd((Mtiles + 3) / 4, Cout / 64);
        gemm_wmma_bf16_kernel<<<ggrd, gblk, 0, stream>>>(
            actbf, wpack, h_buf, Mtiles, K32, Cout);

        // attention scalars + softmax buffer init
        int nth = N * heads;
        node_scores_kernel<<<(nth + 255) / 256, 256, 0, stream>>>(
            h_buf, a_s, a_d, ssrc, sdst, emax, esum, N, heads, d);

        // zero aggregation target
        launch_zero(agg_buf, (long)N * Cout);

        // edge softmax (3 passes)
        int eb = (E + 255) / 256;
        edge_max_kernel<<<eb, 256, 0, stream>>>(src, dst, ssrc, sdst, emax, E, heads);
        edge_sum_kernel<<<eb, 256, 0, stream>>>(src, dst, ssrc, sdst, emax, esum, E, heads);
        edge_aggregate_kernel<<<E, Cout, 0, stream>>>(
            src, dst, ssrc, sdst, emax, esum, h_buf, agg_buf, heads, d);

        // BatchNorm + ELU (+ bf16 conversion for next GEMM)
        launch_zero(stats, 2L * Cout);
        col_stats_kernel<<<1024, Cout, 0, stream>>>(agg_buf, stats, N, Cout);
        long tot = (long)N * Cout;
        bn_elu_kernel<<<(int)((tot + 255) / 256), 256, 0, stream>>>(
            agg_buf, stats, bn_g, bn_b, actbf, bn_f32_out, N, Cout);
    };

    // ---- layer 1: (N,4) -> (N,256), 4 heads.  Pad K 4 -> 32. ----
    {
        long tot = (long)N * 32;
        pad_cvt_kernel<<<(int)((tot + 255) / 256), 256, 0, stream>>>(x, actbf, N, 4, 32);
    }
    run_layer(W1, 4, 32, 256, 4, D, as1, ad1, g1, b1, nullptr);
    // ---- layer 2: (N,256) -> (N,256), 4 heads ----
    run_layer(W2, 256, 256, 256, 4, D, as2, ad2, g2, b2, nullptr);
    // ---- layer 3: (N,256) -> (N,64), 1 head ----
    run_layer(W3, 256, 256, 64, 1, D, as3, ad3, g3, b3, act3);

    // ---- pooling ----
    launch_zero(counts, G);
    launch_zero(psum, (long)G * D);
    launch_zero(pmax, (long)G * D);
    long pel = (long)N * D;
    pool_kernel<<<(int)((pel + 255) / 256), 256, 0, stream>>>(
        act3, batch, psum, pmax, counts, N, D);
    gfeat_kernel<<<(G * 2 * D + 127) / 128, 128, 0, stream>>>(
        psum, pmax, counts, gfeat, G, D);

    // ---- classifier head ----
    classifier_kernel<<<1, 128, 0, stream>>>(
        gfeat, Wc1, bc1, gc1, bnc1, Wc2, bc2, Wc3, bc3, out, G);
}
